// LatentBrainCore_70205535420694
// MI455X (gfx1250) — matrix operand order repaired and checked by reading, not compiled, not run on hardware
//
#include <hip/hip_runtime.h>
#include <math.h>

typedef __bf16 bf16;
typedef __attribute__((ext_vector_type(16))) __bf16 v16bf;
typedef __attribute__((ext_vector_type(8)))  __bf16 v8bf;
typedef __attribute__((ext_vector_type(8)))  float  v8f;

#define DIMX   2048
#define HEADS  16
#define DKK    128
#define NSEQ   4096
#define BATCH  4
#define BH     (BATCH*HEADS)
#define QKVW   6144
#define HALF_D 1024
#define INV_SQRT_DK 0.08838834764831845f

// GEMM tiling: 256 threads = 8 waves (4 m-waves x 2 n-waves).
// Wave tile = 32 x (16*JN).  Block tile = 128 x (32*JN).  K-step = 32.
// LDS strides padded to 40 elems (80B): fragment chunks are 16B-aligned
// contiguous runs -> ds_load_b128.  Double-buffered (ping-pong) LDS with
// register prefetch.  All B fragments are loaded into distinct registers
// before the WMMA burst so ds latency overlaps XDL issue.
#define BM 128
#define BK 32
#define LDA_S 40
#define LDBT_S 40
#define NTHREADS 256

__device__ __forceinline__ v8f wmma_bf16(v16bf a, v16bf b, v8f c) {
  return __builtin_amdgcn_wmma_f32_16x16x32_bf16(false, a, false, b, (short)0, c,
                                                 false, false);
}

// ---- prefetch (global -> regs) and commit (regs -> LDS) -------------------
struct PrefA  { uint4 d0, d1; };                 // 32B of bf16 A row chunk
struct PrefAf { float4 f0, f1, f2, f3; };        // 16 f32 of A row chunk
template<int NV> struct PrefB { v8bf v[NV]; };   // BN_/8 bf16 of one B row

__device__ __forceinline__ PrefA pref_A_bf16(const bf16* A, long lda, int tid) {
  int row = tid >> 1, colb = (tid & 1) << 4;
  const bf16* src = A + (long)row * lda + colb;
  PrefA p;
  p.d0 = *(const uint4*)(src);
  p.d1 = *(const uint4*)(src + 8);
  return p;
}
__device__ __forceinline__ void commit_A_bf16(const PrefA& p, bf16* As, int tid) {
  int row = tid >> 1, colb = (tid & 1) << 4;
  bf16* dst = As + row * LDA_S + colb;        // 16B aligned (row*80 + colb*2)
  *(uint4*)(dst)     = p.d0;
  *(uint4*)(dst + 8) = p.d1;
}

__device__ __forceinline__ PrefAf pref_A_f32(const float* A, long lda, int tid) {
  int row = tid >> 1, colb = (tid & 1) << 4;
  const float* src = A + (long)row * lda + colb;
  PrefAf p;
  p.f0 = *(const float4*)(src);
  p.f1 = *(const float4*)(src + 4);
  p.f2 = *(const float4*)(src + 8);
  p.f3 = *(const float4*)(src + 12);
  return p;
}
__device__ __forceinline__ void commit_A_f32(const PrefAf& p, bf16* As, int tid) {
  int row = tid >> 1, colb = (tid & 1) << 4;
  bf16* dst = As + row * LDA_S + colb;
  float t[16] = { p.f0.x, p.f0.y, p.f0.z, p.f0.w, p.f1.x, p.f1.y, p.f1.z, p.f1.w,
                  p.f2.x, p.f2.y, p.f2.z, p.f2.w, p.f3.x, p.f3.y, p.f3.z, p.f3.w };
#pragma unroll
  for (int c = 0; c < 16; ++c) dst[c] = (bf16)t[c];
}

// B tile stored TRANSPOSED in LDS: Bs[n][k] so fragment reads are contiguous.
template<int BN_>
__device__ __forceinline__ PrefB<BN_ / 64> pref_Bt(const bf16* B, long ldb, int tid) {
  constexpr int E = BN_ / 8, NV = E / 8;
  int k = tid >> 3, n0 = (tid & 7) * E;
  const bf16* src = B + (long)k * ldb + n0;   // contiguous in n
  PrefB<NV> p;
#pragma unroll
  for (int i = 0; i < NV; ++i) p.v[i] = *(const v8bf*)(src + 8 * i);
  return p;
}
template<int BN_>
__device__ __forceinline__ void commit_Bt(const PrefB<BN_ / 64>& p, bf16* Bs, int tid) {
  constexpr int E = BN_ / 8;
  int k = tid >> 3, n0 = (tid & 7) * E;
  bf16* col = Bs + k;
#pragma unroll
  for (int c = 0; c < E; ++c)
    col[(n0 + c) * LDBT_S] = p.v[c >> 3][c & 7];   // transpose scatter
}

// ---- LDS -> WMMA fragment loaders (documented CDNA5 wave32 layouts) -------
// A 16x32: lane L (M=L%16): grp=L>>4 reads elems [grp*8 .. +7] and
// [16+grp*8 .. +7] -> two 16B chunks.
__device__ __forceinline__ v16bf load_fragA(const bf16* As, int mbase, int lane) {
  const bf16* r = As + (mbase + (lane & 15)) * LDA_S + ((lane >> 4) << 3);
  v8bf lo = *(const v8bf*)(r);
  v8bf hi = *(const v8bf*)(r + 16);
  return __builtin_shufflevector(lo, hi, 0, 1, 2, 3, 4, 5, 6, 7,
                                 8, 9, 10, 11, 12, 13, 14, 15);
}

// B 32x16 from transposed LDS: lane L (N=L%16) reads K=[grp*8 .. +7]
// and [16+grp*8 .. +7] at Bs[n][...].
__device__ __forceinline__ v16bf load_fragB(const bf16* Bs, int nbase, int lane) {
  const bf16* r = Bs + (nbase + (lane & 15)) * LDBT_S + ((lane >> 4) << 3);
  v8bf lo = *(const v8bf*)(r);
  v8bf hi = *(const v8bf*)(r + 16);
  return __builtin_shufflevector(lo, hi, 0, 1, 2, 3, 4, 5, 6, 7,
                                 8, 9, 10, 11, 12, 13, 14, 15);
}

template<int NB, int JN>
__device__ __forceinline__ void zero_acc(v8f (&a)[NB][2][JN]) {
#pragma unroll
  for (int n = 0; n < NB; ++n)
#pragma unroll
    for (int i = 0; i < 2; ++i)
#pragma unroll
      for (int j = 0; j < JN; ++j)
#pragma unroll
        for (int e = 0; e < 8; ++e) a[n][i][j][e] = 0.f;
}

// Core: C(128 x 32*JN) accumulate over K.  A row-major (lda), NB B mats (ldb).
// As: 2 x BM*LDA_S.  Bs: 2 x NB*BN_*LDBT_S  (ping-pong).
template<int JN, int NB, bool AF32>
__device__ __forceinline__ void gemm_core(const void* Ap, long lda,
                                          const bf16* const* Bp, long ldb,
                                          int K, bf16* As, bf16* Bs,
                                          v8f acc[][2][JN]) {
  constexpr int BN_ = 32 * JN;
  constexpr int ASZ = BM * LDA_S;
  constexpr int BSZ = NB * BN_ * LDBT_S;
  int tid  = threadIdx.x;
  int lane = tid & 31;
  int wid  = tid >> 5;
  int wm   = (wid >> 1) << 5;        // 0,32,64,96
  int wn   = (wid & 1) * (16 * JN);  // 0 or 16*JN

  // fill tile 0
  if (AF32) { PrefAf p = pref_A_f32((const float*)Ap, lda, tid); commit_A_f32(p, As, tid); }
  else      { PrefA  p = pref_A_bf16((const bf16*)Ap, lda, tid); commit_A_bf16(p, As, tid); }
#pragma unroll
  for (int nb = 0; nb < NB; ++nb) {
    PrefB<BN_ / 64> p = pref_Bt<BN_>(Bp[nb], ldb, tid);
    commit_Bt<BN_>(p, Bs + nb * (BN_ * LDBT_S), tid);
  }
  __syncthreads();

  int cur = 0;
  for (int k0 = 0; k0 < K; k0 += BK) {
    bool hasnext = (k0 + BK) < K;
    // 1) issue next tile's global loads (latency hides behind the WMMAs)
    PrefA  pa;
    PrefAf paf;
    PrefB<BN_ / 64> pb[NB];
    if (hasnext) {
      if (AF32) paf = pref_A_f32((const float*)Ap + k0 + BK, lda, tid);
      else      pa  = pref_A_bf16((const bf16*)Ap + k0 + BK, lda, tid);
#pragma unroll
      for (int nb = 0; nb < NB; ++nb)
        pb[nb] = pref_Bt<BN_>(Bp[nb] + (long)(k0 + BK) * ldb, ldb, tid);
    }
    // 2) compute on current LDS buffer: load ALL fragments into distinct
    //    registers first, then burst the WMMAs (overlaps ds latency).
    const bf16* as = As + cur * ASZ;
    v16bf a0 = load_fragA(as, wm, lane);
    v16bf a1 = load_fragA(as, wm + 16, lane);
    v16bf bfr[NB][JN];
#pragma unroll
    for (int nb = 0; nb < NB; ++nb)
#pragma unroll
      for (int j = 0; j < JN; ++j)
        bfr[nb][j] = load_fragB(Bs + cur * BSZ + nb * (BN_ * LDBT_S),
                                wn + 16 * j, lane);
#pragma unroll
    for (int nb = 0; nb < NB; ++nb)
#pragma unroll
      for (int j = 0; j < JN; ++j) {
        acc[nb][0][j] = wmma_bf16(a0, bfr[nb][j], acc[nb][0][j]);
        acc[nb][1][j] = wmma_bf16(a1, bfr[nb][j], acc[nb][1][j]);
      }
    // 3) commit prefetched tile into the other buffer
    if (hasnext) {
      bf16* an = As + (cur ^ 1) * ASZ;
      bf16* bn = Bs + (cur ^ 1) * BSZ;
      if (AF32) commit_A_f32(paf, an, tid); else commit_A_bf16(pa, an, tid);
#pragma unroll
      for (int nb = 0; nb < NB; ++nb)
        commit_Bt<BN_>(pb[nb], bn + nb * (BN_ * LDBT_S), tid);
    }
    __syncthreads();
    cur ^= 1;
  }
}

// Epilogue iterator: f(row, col, i, j, r) over this thread's C elements.
// C/D 16x16 f32 layout: lane<16 -> N=lane, M=r ; lane>=16 -> M=8+r.
template<int JN, typename F>
__device__ __forceinline__ void epilogue(F f) {
  int lane = threadIdx.x & 31, wid = threadIdx.x >> 5;
  int wm = (wid >> 1) << 5, wn = (wid & 1) * (16 * JN);
  int ln = lane & 15, lh = lane >> 4;
#pragma unroll
  for (int i = 0; i < 2; ++i)
#pragma unroll
    for (int j = 0; j < JN; ++j)
#pragma unroll
      for (int r = 0; r < 8; ++r)
        f(wm + 16 * i + 8 * lh + r, wn + 16 * j + ln, i, j, r);
}

// ===========================================================================
// Stage kernels
// ===========================================================================

// qkv[16384 x 6144] = x[16384 x 2048] @ Wcat[2048 x 6144]   (bf16 out)
__global__ __launch_bounds__(NTHREADS)
void g_qkv(const float* __restrict__ x, const bf16* __restrict__ Wcat,
           bf16* __restrict__ qkv) {
  __shared__ __align__(16) bf16 As[2 * BM * LDA_S];
  __shared__ __align__(16) bf16 Bs[2 * 128 * LDBT_S];
  const float* A = x + (long)blockIdx.x * BM * DIMX;
  const bf16* Bp[1] = { Wcat + (long)blockIdx.y * 128 };
  v8f acc[1][2][4]; zero_acc(acc);
  gemm_core<4, 1, true>(A, DIMX, Bp, QKVW, DIMX, As, Bs, acc);
  long rowbase = (long)blockIdx.x * BM;
  int  colbase = blockIdx.y * 128;
  epilogue<4>([&](int row, int col, int i, int j, int r) {
    qkv[(rowbase + row) * (long)QKVW + colbase + col] = (bf16)acc[0][i][j][r];
  });
}

// per (b,h): res = k_bh[4096x128] @ Wb[h][128x128]; fused k_mod + decay,
// transposed stores: k{fast,slow}T[bh][d][n]
__global__ __launch_bounds__(NTHREADS)
void g_res(const bf16* __restrict__ qkv, const bf16* __restrict__ Wbb,
           const float* __restrict__ fast_slope, const float* __restrict__ slow_slope,
           bf16* __restrict__ kfastT, bf16* __restrict__ kslowT) {
  __shared__ __align__(16) bf16 As[2 * BM * LDA_S];
  __shared__ __align__(16) bf16 Bs[2 * 128 * LDBT_S];
  int bh = blockIdx.z, b = bh >> 4, h = bh & 15;
  const bf16* kbase = qkv + (long)b * NSEQ * QKVW + DIMX + h * DKK;
  const bf16* A = kbase + (long)blockIdx.x * BM * QKVW;
  const bf16* Bp[1] = { Wbb + (long)h * DKK * DKK };
  v8f acc[1][2][4]; zero_acc(acc);
  gemm_core<4, 1, false>(A, QKVW, Bp, DKK, DKK, As, Bs, acc);
  float fs = fast_slope[h], ss = slow_slope[h];
  epilogue<4>([&](int row, int col, int i, int j, int r) {
    int npos = blockIdx.x * BM + row;
    int e    = col;
    float res = acc[0][i][j][r];
    float kv  = (float)kbase[(long)npos * QKVW + e];
    float km  = kv * (1.f + 0.5f * tanhf(res)) * INV_SQRT_DK;
    float iw  = (float)(npos - (NSEQ - 1));
    long  oi  = ((long)bh * DKK + e) * NSEQ + npos;
    kfastT[oi] = (bf16)(km * expf(fs * iw));
    kslowT[oi] = (bf16)(km * expf(ss * iw));
  });
}

// per (b,h): M = kT[128x4096] @ v_bh[4096x128]; M = Mprev*inter + M
__global__ __launch_bounds__(NTHREADS)
void g_mcur(const bf16* __restrict__ kT, const bf16* __restrict__ qkv,
            const float* __restrict__ Mprev, const float* __restrict__ inter,
            float* __restrict__ Mf32, bf16* __restrict__ Mbf) {
  __shared__ __align__(16) bf16 As[2 * BM * LDA_S];
  __shared__ __align__(16) bf16 Bs[2 * 128 * LDBT_S];
  int bh = blockIdx.z, b = bh >> 4, h = bh & 15;
  const bf16* A = kT + (long)bh * DKK * NSEQ;
  const bf16* Bp[1] = { qkv + (long)b * NSEQ * QKVW + 2 * DIMX + h * DKK };
  v8f acc[1][2][4]; zero_acc(acc);
  gemm_core<4, 1, false>(A, NSEQ, Bp, QKVW, NSEQ, As, Bs, acc);
  float iv = inter[h];
  epilogue<4>([&](int row, int col, int i, int j, int r) {
    long idx = ((long)bh * DKK + row) * DKK + col;
    float m = Mprev[idx] * iv + acc[0][i][j][r];
    Mf32[idx] = m;
    Mbf[idx]  = (bf16)m;
  });
}

// per (b,h): Mref = (I+S)[h][128x128] @ M[bh][128x128]
__global__ __launch_bounds__(NTHREADS)
void g_mref(const bf16* __restrict__ IpS, const bf16* __restrict__ Mbf,
            bf16* __restrict__ Mref) {
  __shared__ __align__(16) bf16 As[2 * BM * LDA_S];
  __shared__ __align__(16) bf16 Bs[2 * 128 * LDBT_S];
  int bh = blockIdx.z, h = bh & 15;
  const bf16* A = IpS + (long)h * DKK * DKK;
  const bf16* Bp[1] = { Mbf + (long)bh * DKK * DKK };
  v8f acc[1][2][4]; zero_acc(acc);
  gemm_core<4, 1, false>(A, DKK, Bp, DKK, DKK, As, Bs, acc);
  epilogue<4>([&](int row, int col, int i, int j, int r) {
    Mref[(long)bh * DKK * DKK + (long)row * DKK + col] = (bf16)acc[0][i][j][r];
  });
}

// per (b,h): o_fast = q@Mf, o_slow = q@Ms in one pass; blend with psi -> bf16
// JN=2 (BN=64) keeps the dual accumulator set at 64 VGPRs.
__global__ __launch_bounds__(NTHREADS)
void g_o(const bf16* __restrict__ qkv, const bf16* __restrict__ MrefF,
         const bf16* __restrict__ MrefS, const float* __restrict__ psi,
         bf16* __restrict__ oblend) {
  __shared__ __align__(16) bf16 As[2 * BM * LDA_S];
  __shared__ __align__(16) bf16 Bs[2 * 2 * 64 * LDBT_S];
  int bh = blockIdx.z, b = bh >> 4, h = bh & 15;
  const bf16* A = qkv + (long)b * NSEQ * QKVW + h * DKK +
                  (long)blockIdx.x * BM * QKVW;
  const bf16* Bp[2] = { MrefF + (long)bh * DKK * DKK + blockIdx.y * 64,
                        MrefS + (long)bh * DKK * DKK + blockIdx.y * 64 };
  v8f acc[2][2][2]; zero_acc(acc);
  gemm_core<2, 2, false>(A, QKVW, Bp, DKK, DKK, As, Bs, acc);
  float p = psi[bh];
  epilogue<2>([&](int row, int col, int i, int j, int r) {
    int npos = blockIdx.x * BM + row;
    int dv   = blockIdx.y * 64 + col;
    float o = (1.f - p) * acc[0][i][j][r] + p * acc[1][i][j][r];
    oblend[((long)b * NSEQ + npos) * DIMX + h * DKK + dv] = (bf16)o;
  });
}

// out[16384 x 2048] = o_blend[16384 x 2048] @ Wo[2048 x 2048]  (f32 out)
__global__ __launch_bounds__(NTHREADS)
void g_out(const bf16* __restrict__ ob, const bf16* __restrict__ Wob,
           float* __restrict__ out) {
  __shared__ __align__(16) bf16 As[2 * BM * LDA_S];
  __shared__ __align__(16) bf16 Bs[2 * 128 * LDBT_S];
  const bf16* A = ob + (long)blockIdx.x * BM * DIMX;
  const bf16* Bp[1] = { Wob + (long)blockIdx.y * 128 };
  v8f acc[1][2][4]; zero_acc(acc);
  gemm_core<4, 1, false>(A, DIMX, Bp, DIMX, DIMX, As, Bs, acc);
  epilogue<4>([&](int row, int col, int i, int j, int r) {
    out[((long)blockIdx.x * BM + row) * DIMX + blockIdx.y * 128 + col] =
        acc[0][i][j][r];
  });
}

// ---- prep / small kernels --------------------------------------------------
__global__ void k_cat(const float* __restrict__ Wq, const float* __restrict__ Wk,
                      const float* __restrict__ Wv, bf16* __restrict__ Wcat) {
  long i = (long)blockIdx.x * blockDim.x + threadIdx.x;
  long row = i / QKVW; int c = (int)(i % QKVW);
  float v;
  if (c < DIMX)            v = Wq[row * DIMX + c];
  else if (c < 2 * DIMX)   v = Wk[row * DIMX + (c - DIMX)];
  else                     v = Wv[row * DIMX + (c - 2 * DIMX)];
  Wcat[i] = (bf16)v;
}

__global__ void k_cvt(const float* __restrict__ src, bf16* __restrict__ dst, long n) {
  long i = (long)blockIdx.x * blockDim.x + threadIdx.x;
  if (i < n) dst[i] = (bf16)src[i];
}

__global__ void k_ips(const float* __restrict__ Sf, const float* __restrict__ Ss,
                      bf16* __restrict__ IpSf, bf16* __restrict__ IpSs) {
  int i = blockIdx.x * blockDim.x + threadIdx.x;  // 0..262143
  int e = i & 127, d = (i >> 7) & 127;
  float add = (d == e) ? 1.f : 0.f;
  IpSf[i] = (bf16)(Sf[i] + add);
  IpSs[i] = (bf16)(Ss[i] + add);
}

__global__ void k_xmean(const float* __restrict__ x, float* __restrict__ xmean) {
  int i = blockIdx.x * blockDim.x + threadIdx.x;  // 0..8191
  int b = i >> 11, d = i & 2047;
  const float* p = x + (long)b * NSEQ * DIMX + d;
  float s = 0.f;
  for (int n = 0; n < NSEQ; ++n) s += p[(long)n * DIMX];
  xmean[i] = s * (1.f / NSEQ);
}

__global__ void k_h1(const float* __restrict__ xmean, const float* __restrict__ f1w,
                     const float* __restrict__ f1b, float* __restrict__ h1) {
  int i = blockIdx.x * blockDim.x + threadIdx.x;  // 0..4095
  int b = i >> 10, j = i & 1023;
  const float* xm = xmean + b * DIMX;
  float s = f1b[j];
  for (int d = 0; d < DIMX; ++d) s += xm[d] * f1w[(long)d * HALF_D + j];
  h1[i] = s / (1.f + expf(-s));   // silu
}

__global__ void k_psi(const float* __restrict__ h1, const float* __restrict__ f2w,
                      const float* __restrict__ f2b, float* __restrict__ psi_ws,
                      float* __restrict__ psi_out) {
  int i = threadIdx.x;  // 0..63
  int b = i >> 4, h = i & 15;
  const float* hh = h1 + b * HALF_D;
  float s = f2b[h];
  for (int j = 0; j < HALF_D; ++j) s += hh[j] * f2w[j * HEADS + h];
  float p = 1.f / (1.f + expf(-s));
  psi_ws[i]  = p;
  psi_out[i] = p;
}

// ===========================================================================
extern "C" void kernel_launch(void* const* d_in, const int* in_sizes, int n_in,
                              void* d_out, int out_size, void* d_ws, size_t ws_size,
                              hipStream_t stream) {
  (void)in_sizes; (void)n_in; (void)out_size; (void)ws_size;
  const float* x    = (const float*)d_in[0];
  const float* Mfp  = (const float*)d_in[1];
  const float* Msp  = (const float*)d_in[2];
  const float* Wq   = (const float*)d_in[3];
  const float* Wk   = (const float*)d_in[4];
  const float* Wv   = (const float*)d_in[5];
  const float* Wo   = (const float*)d_in[6];
  const float* Wb   = (const float*)d_in[7];
  const float* fsl  = (const float*)d_in[8];
  const float* ssl  = (const float*)d_in[9];
  const float* inf_ = (const float*)d_in[10];
  const float* ins_ = (const float*)d_in[11];
  const float* Sf   = (const float*)d_in[12];
  const float* Ss   = (const float*)d_in[13];
  const float* f1w  = (const float*)d_in[14];
  const float* f1b  = (const float*)d_in[15];
  const float* f2w  = (const float*)d_in[16];
  const float* f2b  = (const float*)d_in[17];

  char* ws = (char*)d_ws;
  size_t off = 0;
  auto take = [&](size_t bytes) -> char* {
    char* p = ws + off;
    off += (bytes + 255) & ~(size_t)255;
    return p;
  };
  bf16*  qkv   = (bf16*)take((size_t)BATCH * NSEQ * QKVW * 2);   // 192 MB
  bf16*  Wcat  = (bf16*)take((size_t)DIMX * QKVW * 2);           //  24 MB
  bf16*  Wob   = (bf16*)take((size_t)DIMX * DIMX * 2);           //   8 MB
  bf16*  Wbb   = (bf16*)take((size_t)HEADS * DKK * DKK * 2);
  bf16*  kfT   = (bf16*)take((size_t)BH * DKK * NSEQ * 2);       //  64 MB
  bf16*  ksT   = (bf16*)take((size_t)BH * DKK * NSEQ * 2);       //  64 MB
  bf16*  Mtf   = (bf16*)take((size_t)BH * DKK * DKK * 2);
  bf16*  Mts   = (bf16*)take((size_t)BH * DKK * DKK * 2);
  bf16*  Mrf   = (bf16*)take((size_t)BH * DKK * DKK * 2);
  bf16*  Mrs   = (bf16*)take((size_t)BH * DKK * DKK * 2);
  bf16*  IpSf  = (bf16*)take((size_t)HEADS * DKK * DKK * 2);
  bf16*  IpSs  = (bf16*)take((size_t)HEADS * DKK * DKK * 2);
  bf16*  obl   = (bf16*)take((size_t)BATCH * NSEQ * DIMX * 2);   //  64 MB
  float* xmean = (float*)take((size_t)BATCH * DIMX * 4);
  float* h1    = (float*)take((size_t)BATCH * HALF_D * 4);
  float* psiw  = (float*)take((size_t)BH * 4);

  float* out_main = (float*)d_out;
  float* out_Mf   = out_main + (size_t)BATCH * NSEQ * DIMX;
  float* out_Ms   = out_Mf + (size_t)BH * DKK * DKK;
  float* out_psi  = out_Ms + (size_t)BH * DKK * DKK;

  // ---- prep (weights -> bf16, gating MLP) ----
  k_cat  <<<49152, 256, 0, stream>>>(Wq, Wk, Wv, Wcat);
  k_cvt  <<<16384, 256, 0, stream>>>(Wo, Wob, (long)DIMX * DIMX);
  k_cvt  <<<1024,  256, 0, stream>>>(Wb, Wbb, (long)HEADS * DKK * DKK);
  k_ips  <<<1024,  256, 0, stream>>>(Sf, Ss, IpSf, IpSs);
  k_xmean<<<32,    256, 0, stream>>>(x, xmean);
  k_h1   <<<16,    256, 0, stream>>>(xmean, f1w, f1b, h1);
  k_psi  <<<1,      64, 0, stream>>>(h1, f2w, f2b, psiw, out_psi);

  // ---- WMMA pipeline ----
  g_qkv <<<dim3(128, 48),   NTHREADS, 0, stream>>>(x, Wcat, qkv);
  g_res <<<dim3(32, 1, BH), NTHREADS, 0, stream>>>(qkv, Wbb, fsl, ssl, kfT, ksT);
  g_mcur<<<dim3(1, 1, BH),  NTHREADS, 0, stream>>>(kfT, qkv, Mfp, inf_, out_Mf, Mtf);
  g_mcur<<<dim3(1, 1, BH),  NTHREADS, 0, stream>>>(ksT, qkv, Msp, ins_, out_Ms, Mts);
  g_mref<<<dim3(1, 1, BH),  NTHREADS, 0, stream>>>(IpSf, Mtf, Mrf);
  g_mref<<<dim3(1, 1, BH),  NTHREADS, 0, stream>>>(IpSs, Mts, Mrs);
  g_o   <<<dim3(32, 2, BH), NTHREADS, 0, stream>>>(qkv, Mrf, Mrs, psiw, obl);
  g_out <<<dim3(128, 16),   NTHREADS, 0, stream>>>(obl, Wob, out_main);
}
